// LatentSDE_68135361184112
// MI455X (gfx1250) — matrix operand order repaired and checked by reading, not compiled, not run on hardware
//
#include <hip/hip_runtime.h>
#include <hip/hip_bf16.h>

// ---------------- problem constants ----------------
#define TT 128
#define BB 4096
#define DD 32
#define LL 16
#define CC 64
#define HH 128

typedef __attribute__((ext_vector_type(16))) __bf16 v16bf;
typedef __attribute__((ext_vector_type(8)))  float  v8f;
typedef __attribute__((ext_vector_type(4)))  unsigned int u32x4;
typedef __attribute__((ext_vector_type(4)))  float        f32x4;

#define DEV static __device__ __forceinline__

// ---------------- workspace layout ----------------
// swizzled bf16 weights (ushort element offsets)
constexpr size_t SWZ_WIH = 0;       // 32x512   -> 16384
constexpr size_t SWZ_WHH = 16384;   // 128x512  -> 65536
constexpr size_t SWZ_ENC = 81920;   // 128x64   -> 8192
constexpr size_t SWZ_F1Z = 90112;   // 32x128   -> 4096 (rows 0..15 of fW1, zero pad)
constexpr size_t SWZ_F1C = 94208;   // 64x128   -> 8192 (rows 16..79 of fW1)
constexpr size_t SWZ_F2  = 102400;  // 128x128  -> 16384
constexpr size_t SWZ_F3  = 118784;  // 128x16   -> 2048
constexpr size_t SWZ_H1  = 120832;  // 32x128   -> 4096 (hW1 padded)
constexpr size_t SWZ_H2  = 124928;  // 128x128  -> 16384
constexpr size_t SWZ_H3  = 141312;  // 128x16   -> 2048
constexpr size_t SWZ_P   = 143360;  // 32x32    -> 1024 (projW padded)
// byte offsets for the rest
constexpr size_t OFF_CTX_B = 524288;                                   // bf16 ctx (T,B,C)
constexpr size_t OFF_Z0_B  = OFF_CTX_B + (size_t)TT*BB*CC*2;           // f32 z0 (B,L)
constexpr size_t OFF_KL_B  = OFF_Z0_B  + (size_t)BB*LL*4;              // f32 kl  (B)
constexpr size_t OFF_LR_B  = OFF_KL_B  + (size_t)BB*4;                 // f32 logr slots (B/16)
constexpr size_t OFF_LP_B  = OFF_LR_B  + (size_t)(BB/16)*4;            // f32 logp slots (B/16)

// ---------------- small helpers ----------------
// native f32->bf16 (v_cvt_bf16_f32 / v_cvt_pk_bf16_f32, RNE)
DEV unsigned short bfbits(float x) {
  return __builtin_bit_cast(unsigned short, (__bf16)x);
}
DEV float bf2f(unsigned short h) { return __builtin_bit_cast(float, ((unsigned)h) << 16); }
DEV float sigm(float x) { return 1.0f / (1.0f + __expf(-x)); }
DEV v8f splat8(float x) { v8f v = {x,x,x,x,x,x,x,x}; return v; }

union FragU { u32x4 q[2]; v16bf v; };

// B fragment from swizzled weights (global, L2-resident): 2x b128 per lane.
DEV v16bf load_b_swz(const unsigned short* swz, int nt, int kc, int KC, int lane) {
  const unsigned short* p = swz + (((size_t)(nt * KC + kc) * 32 + lane) * 16);
  FragU u;
  u.q[0] = *(const u32x4*)(p);
  u.q[1] = *(const u32x4*)(p + 8);
  return u.v;
}

// A fragment from a row-major bf16 tile (LDS or global). stride in ushorts, mult of 8.
// lane<16: row=lane, k = kbase+{0..7,16..23}; lane>=16: row=lane-16, k = kbase+{8..15,24..31}
DEV v16bf load_a_bf(const unsigned short* tile, int stride, int kbase, int lane) {
  int row = lane & 15, hi = lane >> 4;
  const unsigned short* p = tile + (size_t)row * stride + kbase + hi * 8;
  FragU u;
  u.q[0] = *(const u32x4*)(p);
  u.q[1] = *(const u32x4*)(p + 16);
  return u.v;
}

// A fragment from row-major f32 global (one K=32 chunk), converted to bf16.
DEV v16bf load_a_f32(const float* base, int stride, int lane) {
  int row = lane & 15, hi = lane >> 4;
  const float* p = base + (size_t)row * stride + hi * 8;
  union { f32x4 q[4]; float f[16]; } u;
  u.q[0] = *(const f32x4*)(p);
  u.q[1] = *(const f32x4*)(p + 4);
  u.q[2] = *(const f32x4*)(p + 16);
  u.q[3] = *(const f32x4*)(p + 20);
  v16bf v;
#pragma unroll
  for (int j = 0; j < 16; ++j) v[j] = (__bf16)u.f[j];
  return v;
}

DEV v8f wmma_bf(v16bf a, v16bf b, v8f c) {
  return __builtin_amdgcn_wmma_f32_16x16x32_bf16(false, a, false, b, (short)0, c, false, false);
}

// prefetch a [16 x rowlen] f32/bf16 tile's rows one step ahead (global_prefetch_b8)
DEV void prefetch_rows(const void* base, size_t row_bytes, int lane) {
  // 16 rows; one lane per row (lanes 16..31 prefetch same rows at +64B to cover the line)
  const char* p = (const char*)base + (size_t)(lane & 15) * row_bytes + (lane >> 4) * 64;
  __builtin_prefetch(p, 0, 1);
}

// ---------------- kernel 0: swizzle all weights into ws (bf16, fragment-native) ----------------
DEV void fill_swz(unsigned short* dst, const float* W, int Ksrc, int Kpad, int N,
                  int rowOff, int srcN) {
  int KC = Kpad >> 5;
  int total = (N >> 4) * KC * 512;
  for (int idx = blockIdx.x * blockDim.x + threadIdx.x; idx < total;
       idx += gridDim.x * blockDim.x) {
    int j = idx & 15, lane = (idx >> 4) & 31, t2 = idx >> 9;
    int kc = t2 % KC, nt = t2 / KC;
    int k = kc * 32 + (j & 7) + ((lane >> 4) << 3) + ((j >> 3) << 4);        // B layout: k = 8*laneHi + (j&7) + 16*(j>>3)
    float v = (k < Ksrc) ? W[(size_t)(k + rowOff) * srcN + nt * 16 + (lane & 15)] : 0.0f;
    dst[idx] = bfbits(v);
  }
}

__global__ void swz_weights(const float* wih, const float* whh, const float* encw,
                            const float* fw1, const float* fw2, const float* fw3,
                            const float* hw1, const float* hw2, const float* hw3,
                            const float* pw, unsigned short* ws) {
  fill_swz(ws + SWZ_WIH, wih,  32, 32, 512,  0, 512);
  fill_swz(ws + SWZ_WHH, whh, 128,128, 512,  0, 512);
  fill_swz(ws + SWZ_ENC, encw,128,128,  64,  0,  64);
  fill_swz(ws + SWZ_F1Z, fw1,  16, 32, 128,  0, 128);
  fill_swz(ws + SWZ_F1C, fw1,  64, 64, 128, 16, 128);
  fill_swz(ws + SWZ_F2,  fw2, 128,128, 128,  0, 128);
  fill_swz(ws + SWZ_F3,  fw3, 128,128,  16,  0,  16);
  fill_swz(ws + SWZ_H1,  hw1,  16, 32, 128,  0, 128);
  fill_swz(ws + SWZ_H2,  hw2, 128,128, 128,  0, 128);
  fill_swz(ws + SWZ_H3,  hw3, 128,128,  16,  0,  16);
  fill_swz(ws + SWZ_P,   pw,   16, 32,  32,  0,  32);
}

// ---------------- kernel 1: backward LSTM encoder + ctx + qz0/z0/KL ----------------
// 64 threads = 2 waves; each wave owns a 16-row batch tile, walks t = T-1..0.
__global__ void __launch_bounds__(64) enc_kernel(
    const float* xs, const float* bih, const float* bhh, const float* encb,
    const float* qzw, const float* qzb, const float* eps0,
    const float* pz0m, const float* pz0l,
    const unsigned short* swz, unsigned short* ctxw, float* z0w, float* klw) {
  __shared__ float sBias[512];
  __shared__ float sEncB[64];
  __shared__ unsigned short sH[2][2][16 * 128];  // bf16 h, per-wave double buffer
  __shared__ float sC[2][16 * 128];              // f32 cell state
  __shared__ float sQz[2][16][32];

  int tid = threadIdx.x, w = tid >> 5, lane = tid & 31;
  for (int i = tid; i < 512; i += 64) sBias[i] = bih[i] + bhh[i];
  for (int i = tid; i < 64; i += 64) sEncB[i] = encb[i];
  __syncthreads();

  int b0 = (blockIdx.x * 2 + w) * 16;
  for (int i = lane; i < 2048; i += 32) { sH[w][0][i] = 0; sC[w][i] = 0.0f; }
  int col16 = lane & 15, m0 = (lane >> 4) * 8;
  int p = 0;

  for (int t = TT - 1; t >= 0; --t) {
    // pull next step's x tile toward the WGP while this step computes
    if (t > 0)
      prefetch_rows(xs + ((size_t)(t - 1) * BB + b0) * DD, DD * 4, lane);

    const unsigned short* hin = sH[w][p];
    unsigned short* hout = sH[w][p ^ 1];
    v16bf ax = load_a_f32(xs + ((size_t)t * BB + b0) * DD, DD, lane);
    v16bf ah[4];
#pragma unroll
    for (int kc = 0; kc < 4; ++kc) ah[kc] = load_a_bf(hin, 128, kc * 32, lane);

    for (int ng = 0; ng < 8; ++ng) {        // 16 hidden units per group
      v8f ag[4];
#pragma unroll
      for (int g = 0; g < 4; ++g) {         // i,f,g,o
        int nt = g * 8 + ng;
        v8f acc = splat8(sBias[nt * 16 + col16]);
        acc = wmma_bf(ax, load_b_swz(swz + SWZ_WIH, nt, 0, 1, lane), acc);
#pragma unroll
        for (int kc = 0; kc < 4; ++kc)
          acc = wmma_bf(ah[kc], load_b_swz(swz + SWZ_WHH, nt, kc, 4, lane), acc);
        ag[g] = acc;
      }
      int col = ng * 16 + col16;
#pragma unroll
      for (int r = 0; r < 8; ++r) {
        float co = sC[w][(m0 + r) * 128 + col];
        float iv = sigm(ag[0][r]), fv = sigm(ag[1][r]);
        float gv = tanhf(ag[2][r]), ov = sigm(ag[3][r]);
        float cn = fv * co + iv * gv;
        sC[w][(m0 + r) * 128 + col] = cn;
        hout[(m0 + r) * 128 + col] = bfbits(ov * tanhf(cn));
      }
    }
    // ctx[t] = h @ encW + encb  (bf16 to workspace)
    v16bf ah2[4];
#pragma unroll
    for (int kc = 0; kc < 4; ++kc) ah2[kc] = load_a_bf(hout, 128, kc * 32, lane);
    for (int nt = 0; nt < 4; ++nt) {
      v8f acc = splat8(sEncB[nt * 16 + col16]);
#pragma unroll
      for (int kc = 0; kc < 4; ++kc)
        acc = wmma_bf(ah2[kc], load_b_swz(swz + SWZ_ENC, nt, kc, 4, lane), acc);
#pragma unroll
      for (int r = 0; r < 8; ++r)
        ctxw[((size_t)t * BB + b0 + m0 + r) * CC + nt * 16 + col16] = bfbits(acc[r]);
    }
    p ^= 1;
  }

  // qz0 -> z0, per-row KL (scalar tail, one row per lane<16)
  if (lane < 16) {
    int row = b0 + lane;
    const unsigned short* c0 = ctxw + (size_t)row * CC;  // ctx[0]
    for (int oc = 0; oc < 32; ++oc) {
      float s = qzb[oc];
      for (int k = 0; k < 64; ++k) s += bf2f(c0[k]) * qzw[k * 32 + oc];
      sQz[w][lane][oc] = s;
    }
    float kl = 0.0f;
    for (int l = 0; l < 16; ++l) {
      float qm = sQz[w][lane][l], ql = sQz[w][lane][16 + l];
      float z0v = qm + __expf(ql) * eps0[(size_t)row * LL + l];
      z0w[(size_t)row * LL + l] = z0v;
      float pm = pz0m[l], pl = pz0l[l];
      float e2q = __expf(2.0f * ql), e2p = __expf(2.0f * pl);
      kl += pl - ql + (e2q + (qm - pm) * (qm - pm)) / (2.0f * e2p) - 0.5f;
    }
    klw[row] = kl;
  }
}

// ---------------- kernel 2: Euler-Maruyama scan + projection logp ----------------
__global__ void __launch_bounds__(64) sde_kernel(
    const float* xs, const float* ts, const float* dWt,
    const unsigned short* swz, const unsigned short* ctxw, const float* z0w,
    const float* fb1, const float* fb2, const float* fb3,
    const float* hb1, const float* hb2, const float* hb3,
    const float* gw1, const float* gb1, const float* gw2, const float* gb2,
    const float* pb, float* lrw, float* lpw) {
  __shared__ float sG1[2048], sGb1[2048], sG2[2048], sGb2[16];
  __shared__ float sFb1[128], sFb2[128], sFb3[16];
  __shared__ float sHb1[128], sHb2[128], sHb3[16];
  __shared__ float sPb[32];
  __shared__ unsigned short sZ[2][16 * 32];     // z tile bf16, K padded to 32 (upper half 0)
  __shared__ unsigned short sA1[2][16 * 128];
  __shared__ unsigned short sA2[2][16 * 128];

  int tid = threadIdx.x, w = tid >> 5, lane = tid & 31;
  for (int i = tid; i < 2048; i += 64) { sG1[i] = gw1[i]; sGb1[i] = gb1[i]; sG2[i] = gw2[i]; }
  for (int i = tid; i < 128; i += 64) { sFb1[i] = fb1[i]; sFb2[i] = fb2[i]; sHb1[i] = hb1[i]; sHb2[i] = hb2[i]; }
  if (tid < 16) { sFb3[tid] = fb3[tid]; sHb3[tid] = hb3[tid]; sGb2[tid] = gb2[tid]; }
  if (tid < 32) { sPb[tid] = pb[tid]; }
  __syncthreads();

  int b0 = (blockIdx.x * 2 + w) * 16;
  int col = lane & 15, m0 = (lane >> 4) * 8;
  unsigned short* sZw = sZ[w];
  unsigned short* sA1w = sA1[w];
  unsigned short* sA2w = sA2[w];
  for (int i = lane; i < 512; i += 32) sZw[i] = 0;

  float dt = ts[1] - ts[0];
  float sdt = sqrtf(dt);
  float zreg[8];
#pragma unroll
  for (int r = 0; r < 8; ++r) {
    float z = z0w[(size_t)(b0 + m0 + r) * LL + col];
    zreg[r] = z;
    sZw[(m0 + r) * 32 + col] = bfbits(z);
  }
  float lr_acc = 0.0f, lp_acc = 0.0f;

  // logp at t = 0 (z0)
  {
    v16bf az = load_a_bf(sZw, 32, 0, lane);
    for (int nt = 0; nt < 2; ++nt) {
      v8f acc = splat8(sPb[nt * 16 + col]);
      acc = wmma_bf(az, load_b_swz(swz + SWZ_P, nt, 0, 1, lane), acc);
#pragma unroll
      for (int r = 0; r < 8; ++r) {
        float x = xs[((size_t)0 * BB + b0 + m0 + r) * DD + nt * 16 + col];
        float d = (x - acc[r]) * 100.0f;  // 1/NOISE_STD
        lp_acc += d * d;
      }
    }
  }

  for (int i = 0; i < TT - 1; ++i) {
    // prefetch next step's streamed operands (global_prefetch_b8)
    if (i < TT - 2) {
      prefetch_rows(ctxw + ((size_t)(i + 2) * BB + b0) * CC, CC * 2, lane);
      prefetch_rows(dWt + ((size_t)(i + 1) * BB + b0) * LL, LL * 4, lane);
      prefetch_rows(xs + ((size_t)(i + 2) * BB + b0) * DD, DD * 4, lane);
    }

    const unsigned short* cp = ctxw + ((size_t)(i + 1) * BB + b0) * CC;
    v16bf az  = load_a_bf(sZw, 32, 0, lane);
    v16bf ac0 = load_a_bf(cp, CC, 0, lane);
    v16bf ac1 = load_a_bf(cp, CC, 32, lane);

    // f_net layer1: tanh([z|ctx] @ fW1 + fb1)
    for (int nt = 0; nt < 8; ++nt) {
      v8f acc = splat8(sFb1[nt * 16 + col]);
      acc = wmma_bf(az,  load_b_swz(swz + SWZ_F1Z, nt, 0, 1, lane), acc);
      acc = wmma_bf(ac0, load_b_swz(swz + SWZ_F1C, nt, 0, 2, lane), acc);
      acc = wmma_bf(ac1, load_b_swz(swz + SWZ_F1C, nt, 1, 2, lane), acc);
#pragma unroll
      for (int r = 0; r < 8; ++r) sA1w[(m0 + r) * 128 + nt * 16 + col] = bfbits(tanhf(acc[r]));
    }
    v16bf af[4];
#pragma unroll
    for (int kc = 0; kc < 4; ++kc) af[kc] = load_a_bf(sA1w, 128, kc * 32, lane);
    for (int nt = 0; nt < 8; ++nt) {
      v8f acc = splat8(sFb2[nt * 16 + col]);
#pragma unroll
      for (int kc = 0; kc < 4; ++kc) acc = wmma_bf(af[kc], load_b_swz(swz + SWZ_F2, nt, kc, 4, lane), acc);
#pragma unroll
      for (int r = 0; r < 8; ++r) sA2w[(m0 + r) * 128 + nt * 16 + col] = bfbits(tanhf(acc[r]));
    }
#pragma unroll
    for (int kc = 0; kc < 4; ++kc) af[kc] = load_a_bf(sA2w, 128, kc * 32, lane);
    v8f facc = splat8(sFb3[col]);
#pragma unroll
    for (int kc = 0; kc < 4; ++kc) facc = wmma_bf(af[kc], load_b_swz(swz + SWZ_F3, 0, kc, 4, lane), facc);

    // h_net (reuses sA1/sA2)
    for (int nt = 0; nt < 8; ++nt) {
      v8f acc = splat8(sHb1[nt * 16 + col]);
      acc = wmma_bf(az, load_b_swz(swz + SWZ_H1, nt, 0, 1, lane), acc);
#pragma unroll
      for (int r = 0; r < 8; ++r) sA1w[(m0 + r) * 128 + nt * 16 + col] = bfbits(tanhf(acc[r]));
    }
#pragma unroll
    for (int kc = 0; kc < 4; ++kc) af[kc] = load_a_bf(sA1w, 128, kc * 32, lane);
    for (int nt = 0; nt < 8; ++nt) {
      v8f acc = splat8(sHb2[nt * 16 + col]);
#pragma unroll
      for (int kc = 0; kc < 4; ++kc) acc = wmma_bf(af[kc], load_b_swz(swz + SWZ_H2, nt, kc, 4, lane), acc);
#pragma unroll
      for (int r = 0; r < 8; ++r) sA2w[(m0 + r) * 128 + nt * 16 + col] = bfbits(tanhf(acc[r]));
    }
#pragma unroll
    for (int kc = 0; kc < 4; ++kc) af[kc] = load_a_bf(sA2w, 128, kc * 32, lane);
    v8f hacc = splat8(sHb3[col]);
#pragma unroll
    for (int kc = 0; kc < 4; ++kc) hacc = wmma_bf(af[kc], load_b_swz(swz + SWZ_H3, 0, kc, 4, lane), hacc);

    // per-dim g nets (elementwise in accumulator ownership)
    float gs[8] = {0, 0, 0, 0, 0, 0, 0, 0};
    for (int hh = 0; hh < 128; ++hh) {
      float w1 = sG1[col * 128 + hh], b1 = sGb1[col * 128 + hh], w2 = sG2[col * 128 + hh];
#pragma unroll
      for (int r = 0; r < 8; ++r) gs[r] += tanhf(fmaf(zreg[r], w1, b1)) * w2;
    }
    float gb2c = sGb2[col];
#pragma unroll
    for (int r = 0; r < 8; ++r) {
      float gg = sigm(gs[r] + gb2c);
      float fv = facc[r], hv = hacc[r];
      float u = (fv - hv) / gg;
      lr_acc += u * u;
      float dw = dWt[((size_t)i * BB + b0 + m0 + r) * LL + col] * sdt;
      zreg[r] += fv * dt + gg * dw;
      sZw[(m0 + r) * 32 + col] = bfbits(zreg[r]);
    }

    // logp at t = i+1
    v16bf az2 = load_a_bf(sZw, 32, 0, lane);
    for (int nt = 0; nt < 2; ++nt) {
      v8f acc = splat8(sPb[nt * 16 + col]);
      acc = wmma_bf(az2, load_b_swz(swz + SWZ_P, nt, 0, 1, lane), acc);
#pragma unroll
      for (int r = 0; r < 8; ++r) {
        float x = xs[((size_t)(i + 1) * BB + b0 + m0 + r) * DD + nt * 16 + col];
        float d = (x - acc[r]) * 100.0f;
        lp_acc += d * d;
      }
    }
  }

  // wave reduction -> one slot per wave
#pragma unroll
  for (int m = 16; m >= 1; m >>= 1) {
    lr_acc += __shfl_xor(lr_acc, m, 32);
    lp_acc += __shfl_xor(lp_acc, m, 32);
  }
  if (lane == 0) {
    int slot = blockIdx.x * 2 + w;
    lrw[slot] = 0.5f * dt * lr_acc;   // sum over steps/rows/dims of 0.5*u^2*dt
    lpw[slot] = -0.5f * lp_acc;       // sum of -0.5*d^2
  }
}

// ---------------- kernel 3: final reduction to the two scalars ----------------
__global__ void reduce_kernel(const float* klw, const float* lrw, const float* lpw, float* out) {
  __shared__ float r0[256], r1[256], r2[256];
  int tid = threadIdx.x;
  float skl = 0.0f;
  for (int i = tid; i < BB; i += 256) skl += klw[i];
  float slr = lrw[tid];   // BB/16 == 256 slots
  float slp = lpw[tid];
  r0[tid] = skl; r1[tid] = slr; r2[tid] = slp;
  __syncthreads();
  for (int s = 128; s > 0; s >>= 1) {
    if (tid < s) { r0[tid] += r0[tid + s]; r1[tid] += r1[tid + s]; r2[tid] += r2[tid + s]; }
    __syncthreads();
  }
  if (tid == 0) {
    const float invB = 1.0f / (float)BB;
    // per-element constant: -ln(0.01) - 0.5*ln(2*pi) = 3.6862316...
    const float cst = (float)(TT * DD) * 3.6862316527834183f;
    out[0] = r2[0] * invB + cst;           // log_pxs
    out[1] = r0[0] * invB + r1[0] * invB;  // logqp0 + logqp_path
  }
}

// ---------------- host launcher ----------------
extern "C" void kernel_launch(void* const* d_in, const int* in_sizes, int n_in,
                              void* d_out, int out_size, void* d_ws, size_t ws_size,
                              hipStream_t stream) {
  (void)in_sizes; (void)n_in; (void)out_size; (void)ws_size;
  const float* xs   = (const float*)d_in[0];
  const float* ts   = (const float*)d_in[1];
  const float* eps0 = (const float*)d_in[2];
  const float* dW   = (const float*)d_in[3];
  const float* wih  = (const float*)d_in[4];
  const float* whh  = (const float*)d_in[5];
  const float* bih  = (const float*)d_in[6];
  const float* bhh  = (const float*)d_in[7];
  const float* encw = (const float*)d_in[8];
  const float* encb = (const float*)d_in[9];
  const float* qzw  = (const float*)d_in[10];
  const float* qzb  = (const float*)d_in[11];
  const float* fw1  = (const float*)d_in[12];
  const float* fb1  = (const float*)d_in[13];
  const float* fw2  = (const float*)d_in[14];
  const float* fb2  = (const float*)d_in[15];
  const float* fw3  = (const float*)d_in[16];
  const float* fb3  = (const float*)d_in[17];
  const float* hw1  = (const float*)d_in[18];
  const float* hb1  = (const float*)d_in[19];
  const float* hw2  = (const float*)d_in[20];
  const float* hb2  = (const float*)d_in[21];
  const float* hw3  = (const float*)d_in[22];
  const float* hb3  = (const float*)d_in[23];
  const float* gw1  = (const float*)d_in[24];
  const float* gb1  = (const float*)d_in[25];
  const float* gw2  = (const float*)d_in[26];
  const float* gb2  = (const float*)d_in[27];
  const float* pw   = (const float*)d_in[28];
  const float* pb   = (const float*)d_in[29];
  const float* pz0m = (const float*)d_in[30];
  const float* pz0l = (const float*)d_in[31];

  unsigned short* swz = (unsigned short*)d_ws;
  char* wsb = (char*)d_ws;
  unsigned short* ctxw = (unsigned short*)(wsb + OFF_CTX_B);
  float* z0w = (float*)(wsb + OFF_Z0_B);
  float* klw = (float*)(wsb + OFF_KL_B);
  float* lrw = (float*)(wsb + OFF_LR_B);
  float* lpw = (float*)(wsb + OFF_LP_B);
  float* out = (float*)d_out;

  swz_weights<<<64, 256, 0, stream>>>(wih, whh, encw, fw1, fw2, fw3, hw1, hw2, hw3, pw, swz);
  enc_kernel<<<BB / 32, 64, 0, stream>>>(xs, bih, bhh, encb, qzw, qzb, eps0, pz0m, pz0l,
                                         swz, ctxw, z0w, klw);
  sde_kernel<<<BB / 32, 64, 0, stream>>>(xs, ts, dW, swz, ctxw, z0w,
                                         fb1, fb2, fb3, hb1, hb2, hb3,
                                         gw1, gb1, gw2, gb2, pb, lrw, lpw);
  reduce_kernel<<<1, 256, 0, stream>>>(klw, lrw, lpw, out);
}
